// T5Gemma2DecoderAttention_39582418599958
// MI455X (gfx1250) — matrix operand order repaired and checked by reading, not compiled, not run on hardware
//
#include <hip/hip_runtime.h>
#include <cstdint>
#include <cstddef>

typedef __attribute__((ext_vector_type(16))) __bf16 v16bf;
typedef __attribute__((ext_vector_type(8)))  __bf16 v8bf;
typedef __attribute__((ext_vector_type(8)))  float  v8f;

#define DEVINL __device__ __forceinline__

constexpr int T_   = 2048;
constexpr int H_   = 2048;
constexpr int NH_  = 8;
constexpr int NKV_ = 4;
constexpr int HD_  = 256;
constexpr int B_   = 4;
constexpr int S_   = T_ / B_;                 // 512
constexpr int QKVN = (NH_ + 2 * NKV_) * HD_;  // 4096
constexpr float SCALING_ = 0.0625f;           // 256^-0.5
constexpr float EPS_ = 1e-6f;

DEVINL v8f v8f_zero() {
  v8f z = {0.f, 0.f, 0.f, 0.f, 0.f, 0.f, 0.f, 0.f};
  return z;
}

DEVINL v16bf ld_frag(const __bf16* p0, const __bf16* p1) {
  union { v16bf v; v8bf h[2]; } u;
  u.h[0] = *(const v8bf*)p0;
  u.h[1] = *(const v8bf*)p1;
  return u.v;
}

DEVINL v8f wmma_bf16(v16bf a, v16bf b, v8f c) {
  // D = A(16x32 bf16) * B(32x16 bf16) + C(16x16 f32)
  return __builtin_amdgcn_wmma_f32_16x16x32_bf16(false, a, false, b,
                                                 (short)0, c, false, false);
}

// Async copy 16 bytes from global (saddr base + 32-bit voffset) into LDS.
// GVS mode: mem_addr = SGPR_base + VGPR_offset; VDST = LDS byte address.
DEVINL void async_b128(unsigned lds_addr, unsigned goff, const void* base) {
  asm volatile("global_load_async_to_lds_b128 %0, %1, %2"
               :: "v"(lds_addr), "v"(goff),
                  "s"((unsigned long long)(uintptr_t)base)
               : "memory");
}
DEVINL void wait_async0() {
  asm volatile("s_wait_asynccnt 0" ::: "memory");
}
DEVINL unsigned lds_off(const __bf16* p) {
  return (unsigned)(uintptr_t)p;   // low 32 bits of generic addr == LDS addr
}

// ---------------------------------------------------------------------------
// Generic GEMM: C(f32, MxN) = A(f32, MxK) * B(f32, KxN), bf16 WMMA compute.
// Tile: 128x64, K-step 32, 256 threads (8 waves), double-buffered LDS.
// ---------------------------------------------------------------------------
__global__ __launch_bounds__(256) void gemm_bf16_k(
    const float* __restrict__ A, int lda,
    const float* __restrict__ B, int ldb,
    float* __restrict__ C, int ldc,
    int K)
{
  __shared__ __attribute__((aligned(32))) __bf16 sA[2][128 * 32];
  __shared__ __attribute__((aligned(32))) __bf16 sB[2][64 * 32];   // [n][k]

  const int tid  = threadIdx.x;
  const int lane = tid & 31;
  const int wave = tid >> 5;
  const int ln   = lane & 15;
  const int half = lane >> 4;
  const int bn0  = blockIdx.x * 64;
  const int am0  = blockIdx.y * 128;

  v8f acc[4];
#pragma unroll
  for (int i = 0; i < 4; ++i) acc[i] = v8f_zero();

  const int arow = tid >> 1;            // 0..127
  const int akc  = (tid & 1) * 16;      // 0 or 16
  const int bk   = tid >> 3;            // 0..31
  const int bn   = (tid & 7) * 8;       // 0..56

  auto stageA = [&](int buf, int k0) {
    const float* src = &A[(size_t)(am0 + arow) * lda + k0 + akc];
#pragma unroll
    for (int e = 0; e < 16; ++e) sA[buf][arow * 32 + akc + e] = (__bf16)src[e];
  };
  auto stageB = [&](int buf, int k0) {
    const float* src = &B[(size_t)(k0 + bk) * ldb + bn0 + bn];
#pragma unroll
    for (int e = 0; e < 8; ++e) sB[buf][(bn + e) * 32 + bk] = (__bf16)src[e];
  };

  stageA(0, 0);
  stageB(0, 0);
  int cur = 0;

  for (int k0 = 0; k0 < K; k0 += 32) {
    __syncthreads();
    if (k0 + 32 < K) {
      stageA(cur ^ 1, k0 + 32);
      stageB(cur ^ 1, k0 + 32);
      if (k0 + 64 < K) {
        __builtin_prefetch(&A[(size_t)(am0 + arow) * lda + k0 + 64 + akc], 0, 1);
        __builtin_prefetch(&B[(size_t)(k0 + 64 + bk) * ldb + bn0 + bn], 0, 1);
      }
    }
    // compute from buffer `cur`; preload all fragments, then WMMA chain
    const int m = wave * 16 + ln;
    const int akoff = half ? 8 : 0;
    v16bf afrag = ld_frag(&sA[cur][m * 32 + akoff],
                          &sA[cur][m * 32 + akoff + 16]);
    v16bf bfrag[4];
#pragma unroll
    for (int nt = 0; nt < 4; ++nt) {
      const int n = nt * 16 + ln;
      const int bkoff = half ? 8 : 0;
      bfrag[nt] = ld_frag(&sB[cur][n * 32 + bkoff],
                          &sB[cur][n * 32 + bkoff + 16]);
    }
#pragma unroll
    for (int nt = 0; nt < 4; ++nt) acc[nt] = wmma_bf16(afrag, bfrag[nt], acc[nt]);
    cur ^= 1;
  }

  // C layout: VGPR r, lanes 0-15 -> (M=r, N=lane); lanes 16-31 -> (M=8+r)
#pragma unroll
  for (int nt = 0; nt < 4; ++nt) {
#pragma unroll
    for (int r = 0; r < 8; ++r) {
      const int row = am0 + wave * 16 + (half ? 8 + r : r);
      const int col = bn0 + nt * 16 + ln;
      C[(size_t)row * ldc + col] = acc[nt][r];
    }
  }
}

// ---------------------------------------------------------------------------
// RMSNorm (+1 weight) + RoPE; emits bf16 operands for the attention kernels.
// ---------------------------------------------------------------------------
__global__ __launch_bounds__(256) void normrope_k(
    const float* __restrict__ qkv,    // [T, 4096]
    const float* __restrict__ enckv,  // [T, 2048] (enc K|V)
    const float* __restrict__ cosT,   // [T, 256]
    const float* __restrict__ sinT,   // [T, 256]
    const float* __restrict__ qnw, const float* __restrict__ knw,
    __bf16* __restrict__ qr, __bf16* __restrict__ qc,
    __bf16* __restrict__ kr, __bf16* __restrict__ vb,
    __bf16* __restrict__ kenc, __bf16* __restrict__ venc)
{
  const int t = blockIdx.x, d = threadIdx.x;
  __shared__ float red[256];
  __shared__ float xrow[256];

  const float cs = cosT[t * HD_ + d];
  const float sn = sinT[t * HD_ + d];
  const float* base  = qkv + (size_t)t * QKVN;
  const float* ebase = enckv + (size_t)t * (2 * NKV_ * HD_);

  for (int h = 0; h < NH_; ++h) {
    const float x = base[h * HD_ + d];
    red[d] = x * x; __syncthreads();
    for (int s2 = 128; s2 > 0; s2 >>= 1) {
      if (d < s2) red[d] += red[d + s2];
      __syncthreads();
    }
    const float inv = rsqrtf(red[0] * (1.0f / HD_) + EPS_);
    const float xn = x * inv * (1.f + qnw[d]);
    xrow[d] = xn; __syncthreads();
    const float rot = (d < 128) ? -xrow[d + 128] : xrow[d - 128];
    const size_t o = ((size_t)t * NH_ + h) * HD_ + d;
    qr[o] = (__bf16)(xn * cs + rot * sn);
    qc[o] = (__bf16)(xn * SCALING_);
    __syncthreads();
  }

  for (int h = 0; h < NKV_; ++h) {
    const float x = base[NH_ * HD_ + h * HD_ + d];
    red[d] = x * x; __syncthreads();
    for (int s2 = 128; s2 > 0; s2 >>= 1) {
      if (d < s2) red[d] += red[d + s2];
      __syncthreads();
    }
    const float inv = rsqrtf(red[0] * (1.0f / HD_) + EPS_);
    const float xn = x * inv * (1.f + knw[d]);
    xrow[d] = xn; __syncthreads();
    const float rot = (d < 128) ? -xrow[d + 128] : xrow[d - 128];
    const size_t o = ((size_t)t * NKV_ + h) * HD_ + d;
    kr[o] = (__bf16)(xn * cs + rot * sn);
    __syncthreads();

    vb[o] = (__bf16)base[(NH_ + NKV_) * HD_ + h * HD_ + d];

    const float xe = ebase[h * HD_ + d];
    red[d] = xe * xe; __syncthreads();
    for (int s2 = 128; s2 > 0; s2 >>= 1) {
      if (d < s2) red[d] += red[d + s2];
      __syncthreads();
    }
    const float inve = rsqrtf(red[0] * (1.0f / HD_) + EPS_);
    kenc[o] = (__bf16)(xe * inve * (1.f + knw[d]));
    __syncthreads();

    venc[o] = (__bf16)ebase[NKV_ * HD_ + h * HD_ + d];
  }
}

// ---------------------------------------------------------------------------
// Flash-style attention. Block = 64 threads (2 waves), grid (S/32, NH, B).
// Q/K tiles staged with GLOBAL_LOAD_ASYNC_TO_LDS_B128 (ASYNCcnt); V staged
// manually because it is transposed on the way in.
// ---------------------------------------------------------------------------
template <bool CAUSAL, bool ACCUM>
__global__ __launch_bounds__(64) void attn_k(
    const __bf16* __restrict__ Qb,  // [T, NH, HD]
    const __bf16* __restrict__ Kb,  // [T, NKV, HD]
    const __bf16* __restrict__ Vb,  // [T, NKV, HD]
    float* __restrict__ Ob,         // [T, NH*HD]
    float scoreScale)
{
  __shared__ __attribute__((aligned(32))) __bf16 sQ[32 * 256];
  __shared__ __attribute__((aligned(32))) __bf16 sK[32 * 256];   // [key][d]
  __shared__ __attribute__((aligned(32))) __bf16 sVT[256 * 32];  // [d][key]
  __shared__ __attribute__((aligned(32))) __bf16 sP[2][16 * 32]; // per-wave

  const int tid  = threadIdx.x;
  const int lane = tid & 31;
  const int wave = tid >> 5;
  const int ln   = lane & 15;
  const int half = lane >> 4;
  const int qt = blockIdx.x;
  const int h  = blockIdx.y;
  const int b  = blockIdx.z;
  const int kvh = h >> 1;                 // n_rep = 2
  const int qTok = b * S_ + qt * 32;

  const int srow = tid >> 1;              // 0..31 staging row
  const int sc0  = (tid & 1) * 128;       // staging column base

  // stage Q tile (32 x 256) via async copies
  {
    const __bf16* qbase = Qb + ((size_t)qTok * NH_ + h) * HD_;
#pragma unroll
    for (int j8 = 0; j8 < 16; ++j8) {
      const int c = sc0 + j8 * 8;
      async_b128(lds_off(&sQ[srow * 256 + c]),
                 (unsigned)((srow * (NH_ * HD_) + c) * 2), qbase);
    }
  }

  v8f oacc[16];
#pragma unroll
  for (int i = 0; i < 16; ++i) oacc[i] = v8f_zero();
  float mrow[8], lrow[8];
#pragma unroll
  for (int r = 0; r < 8; ++r) { mrow[r] = -1e30f; lrow[r] = 0.f; }

  const int jcount = CAUSAL ? (qt + 1) : (S_ / 32);

  for (int j = 0; j < jcount; ++j) {
    const int kTok = b * S_ + j * 32;
    __syncthreads();   // previous iteration's LDS reads done
    // stage K (row-major, async)
    {
      const __bf16* kbase = Kb + ((size_t)kTok * NKV_ + kvh) * HD_;
#pragma unroll
      for (int j8 = 0; j8 < 16; ++j8) {
        const int c = sc0 + j8 * 8;
        async_b128(lds_off(&sK[srow * 256 + c]),
                   (unsigned)((srow * (NKV_ * HD_) + c) * 2), kbase);
      }
      // stage V transposed (manual; needs shuffle during copy)
      const __bf16* vsrc = &Vb[(((size_t)(kTok + srow)) * NKV_ + kvh) * HD_ + sc0];
#pragma unroll
      for (int j8 = 0; j8 < 16; ++j8) {
        v8bf vv = *(const v8bf*)(vsrc + j8 * 8);
#pragma unroll
        for (int e = 0; e < 8; ++e)
          sVT[(sc0 + j8 * 8 + e) * 32 + srow] = vv[e];
      }
    }
    wait_async0();
    __syncthreads();

    // S = Q * K^T : wave's 16 rows x 32 keys, K-dim = 256 in 8 steps
    v8f sacc[2];
    sacc[0] = v8f_zero(); sacc[1] = v8f_zero();
#pragma unroll
    for (int kk = 0; kk < 8; ++kk) {
      const int m = wave * 16 + ln;
      const int akoff = kk * 32 + (half ? 8 : 0);
      v16bf aq = ld_frag(&sQ[m * 256 + akoff], &sQ[m * 256 + akoff + 16]);
      const int bkoff = kk * 32 + (half ? 8 : 0);
      v16bf bk0 = ld_frag(&sK[(0 * 16 + ln) * 256 + bkoff],
                          &sK[(0 * 16 + ln) * 256 + bkoff + 16]);
      v16bf bk1 = ld_frag(&sK[(1 * 16 + ln) * 256 + bkoff],
                          &sK[(1 * 16 + ln) * 256 + bkoff + 16]);
      sacc[0] = wmma_bf16(aq, bk0, sacc[0]);
      sacc[1] = wmma_bf16(aq, bk1, sacc[1]);
    }

    // online softmax; each 16-lane half owns 8 rows (C layout)
#pragma unroll
    for (int r = 0; r < 8; ++r) {
      const int lr = half ? 8 + r : r;              // local row in wave tile
      float s0 = sacc[0][r] * scoreScale;
      float s1 = sacc[1][r] * scoreScale;
      if (CAUSAL) {
        const int qpos = qt * 32 + wave * 16 + lr;
        if (j * 32 + ln      > qpos) s0 = -1e30f;
        if (j * 32 + 16 + ln > qpos) s1 = -1e30f;
      }
      float mloc = fmaxf(s0, s1);
#pragma unroll
      for (int off = 8; off >= 1; off >>= 1)
        mloc = fmaxf(mloc, __shfl_xor(mloc, off, 32));
      const float mnew = fmaxf(mrow[r], mloc);
      const float corr = __expf(mrow[r] - mnew);
      const float p0 = __expf(s0 - mnew);
      const float p1 = __expf(s1 - mnew);
      float rs = p0 + p1;
#pragma unroll
      for (int off = 8; off >= 1; off >>= 1)
        rs += __shfl_xor(rs, off, 32);
      lrow[r] = lrow[r] * corr + rs;
      mrow[r] = mnew;
#pragma unroll
      for (int nt = 0; nt < 16; ++nt) oacc[nt][r] *= corr;
      sP[wave][lr * 32 + ln]      = (__bf16)p0;
      sP[wave][lr * 32 + 16 + ln] = (__bf16)p1;
    }

    // O += P(16x32) * V(32x256); wave-local LDS round-trip, in-order DS ops
    {
      const int m = ln;
      const int akoff = half ? 8 : 0;
      v16bf ap = ld_frag(&sP[wave][m * 32 + akoff],
                         &sP[wave][m * 32 + akoff + 16]);
#pragma unroll
      for (int g = 0; g < 4; ++g) {
        v16bf bv[4];
#pragma unroll
        for (int q4 = 0; q4 < 4; ++q4) {
          const int n = (g * 4 + q4) * 16 + ln;     // output dim
          const int bkoff = half ? 8 : 0;
          bv[q4] = ld_frag(&sVT[n * 32 + bkoff], &sVT[n * 32 + bkoff + 16]);
        }
#pragma unroll
        for (int q4 = 0; q4 < 4; ++q4)
          oacc[g * 4 + q4] = wmma_bf16(ap, bv[q4], oacc[g * 4 + q4]);
      }
    }
  }

  // normalize and write (or accumulate onto) output
#pragma unroll
  for (int r = 0; r < 8; ++r) {
    const int lr = half ? 8 + r : r;
    const float inv = 1.0f / lrow[r];
    const size_t row = (size_t)(qTok + wave * 16 + lr);
#pragma unroll
    for (int nt = 0; nt < 16; ++nt) {
      const size_t idx = row * (size_t)(NH_ * HD_) + h * HD_ + nt * 16 + ln;
      const float val = oacc[nt][r] * inv;
      if (ACCUM) Ob[idx] += val;
      else       Ob[idx]  = val;
    }
  }
}

// ---------------------------------------------------------------------------
extern "C" void kernel_launch(void* const* d_in, const int* in_sizes, int n_in,
                              void* d_out, int out_size, void* d_ws,
                              size_t ws_size, hipStream_t stream) {
  (void)in_sizes; (void)n_in; (void)out_size; (void)ws_size;
  const float* hs   = (const float*)d_in[0];
  const float* ehs  = (const float*)d_in[1];
  const float* cosT = (const float*)d_in[2];
  const float* sinT = (const float*)d_in[3];
  const float* wqkv = (const float*)d_in[4];
  const float* wo   = (const float*)d_in[5];
  const float* qnw  = (const float*)d_in[6];
  const float* knw  = (const float*)d_in[7];
  float* out = (float*)d_out;

  char* p = (char*)d_ws;
  auto alloc = [&](size_t bytes) {
    void* r = (void*)p;
    p += (bytes + 255) & ~(size_t)255;
    return r;
  };
  float*  qkv_f   = (float*)alloc((size_t)T_ * QKVN * 4);          // 32 MB
  float*  enckv_f = (float*)alloc((size_t)T_ * 2 * NKV_ * HD_ * 4);// 16 MB
  __bf16* qr      = (__bf16*)alloc((size_t)T_ * NH_ * HD_ * 2);    //  8 MB
  __bf16* qc      = (__bf16*)alloc((size_t)T_ * NH_ * HD_ * 2);    //  8 MB
  __bf16* krb     = (__bf16*)alloc((size_t)T_ * NKV_ * HD_ * 2);   //  4 MB
  __bf16* vbuf    = (__bf16*)alloc((size_t)T_ * NKV_ * HD_ * 2);   //  4 MB
  __bf16* kenc    = (__bf16*)alloc((size_t)T_ * NKV_ * HD_ * 2);   //  4 MB
  __bf16* venc    = (__bf16*)alloc((size_t)T_ * NKV_ * HD_ * 2);   //  4 MB
  float*  attn_f  = (float*)alloc((size_t)T_ * NH_ * HD_ * 4);     // 16 MB

  // 1) QKV projection: [T,H] x [H,4096]
  gemm_bf16_k<<<dim3(QKVN / 64, T_ / 128), 256, 0, stream>>>(
      hs, H_, wqkv, QKVN, qkv_f, QKVN, H_);
  // 2) Encoder K/V projection: [T,H] x [H,2048] (K/V columns of w_qkv)
  gemm_bf16_k<<<dim3((2 * NKV_ * HD_) / 64, T_ / 128), 256, 0, stream>>>(
      ehs, H_, wqkv + NH_ * HD_, QKVN, enckv_f, 2 * NKV_ * HD_, H_);
  // 3) RMSNorm + RoPE -> bf16 attention operands
  normrope_k<<<T_, 256, 0, stream>>>(qkv_f, enckv_f, cosT, sinT, qnw, knw,
                                     qr, qc, krb, vbuf, kenc, venc);
  // 4) Self-attention (causal), writes attn_f
  attn_k<true, false><<<dim3(S_ / 32, NH_, B_), 64, 0, stream>>>(
      qr, krb, vbuf, attn_f, SCALING_);
  // 5) Cross-attention (per-batch full), accumulates onto attn_f
  attn_k<false, true><<<dim3(S_ / 32, NH_, B_), 64, 0, stream>>>(
      qc, kenc, venc, attn_f, 1.0f);
  // 6) Output projection: [T,2048] x [2048,2048] -> d_out (f32)
  gemm_bf16_k<<<dim3(H_ / 64, T_ / 128), 256, 0, stream>>>(
      attn_f, NH_ * HD_, wo, H_, out, H_, NH_ * HD_);
}
